// CrossAttention_8083128451106
// MI455X (gfx1250) — matrix operand order repaired and checked by reading, compiled, not run on hardware
//
#include <hip/hip_runtime.h>

// ---------------------------------------------------------------------------
// CDNA5 (gfx1250) cross-attention, all matmuls via v_wmma_f32_16x16x32_bf16.
// GEMMs: 128x128 LDS-staged, double-buffered, async global->LDS when available.
// ---------------------------------------------------------------------------

typedef __attribute__((ext_vector_type(16))) __bf16 v16bf;
typedef __attribute__((ext_vector_type(8)))  __bf16 v8bf;
typedef __attribute__((ext_vector_type(8)))  float  v8f;
typedef __attribute__((__vector_size__(16))) int v4i_vs;   // matches builtin param

union Frag16 {            // one WMMA 16x16x32 bf16 A/B operand (8 VGPRs/lane)
    v16bf v;
    v8bf  h[2];
    unsigned short s[16];
};

__device__ __forceinline__ unsigned short f2bf(float f) {
    unsigned int u = __builtin_bit_cast(unsigned int, f);
    u += 0x7fffu + ((u >> 16) & 1u);           // round-to-nearest-even
    return (unsigned short)(u >> 16);
}
__device__ __forceinline__ float bf2f(unsigned short s) {
    return __builtin_bit_cast(float, ((unsigned int)s) << 16);
}

#define WMMA_BF16(Afrag, Bfrag, Cacc) \
    __builtin_amdgcn_wmma_f32_16x16x32_bf16(false, (Afrag), false, (Bfrag), (short)0, (Cacc), false, false)

// ---- async global->LDS copy (ASYNCcnt path), with sync fallback ------------
#if defined(__has_builtin)
#if __has_builtin(__builtin_amdgcn_global_load_async_to_lds_b128)
#define HAVE_ASYNC_LDS 1
#endif
#endif
#ifndef HAVE_ASYNC_LDS
#define HAVE_ASYNC_LDS 0
#endif

__device__ __forceinline__ void copy32B_g2l(const unsigned short* g, unsigned short* l) {
#if HAVE_ASYNC_LDS
    __builtin_amdgcn_global_load_async_to_lds_b128(
        (__attribute__((address_space(1))) v4i_vs*)(g),
        (__attribute__((address_space(3))) v4i_vs*)(l), 0, 0);
    __builtin_amdgcn_global_load_async_to_lds_b128(
        (__attribute__((address_space(1))) v4i_vs*)(g + 8),
        (__attribute__((address_space(3))) v4i_vs*)(l + 8), 0, 0);
#else
    *(v16bf*)l = *(const v16bf*)g;
#endif
}

__device__ __forceinline__ void wait_async_all() {
#if HAVE_ASYNC_LDS
#if __has_builtin(__builtin_amdgcn_s_wait_asynccnt)
    __builtin_amdgcn_s_wait_asynccnt(0);
#else
    asm volatile("s_wait_asynccnt 0x0" ::: "memory");
#endif
#endif
}

// ---------------------------------------------------------------------------
// Weight convert + transpose: W[K,N] f32 -> Wt[N,K] bf16
// ---------------------------------------------------------------------------
__global__ __launch_bounds__(256) void cvt_transpose_kernel(
    const float* __restrict__ W, unsigned short* __restrict__ Wt, int K, int N) {
    int idx = blockIdx.x * 256 + threadIdx.x;
    if (idx >= K * N) return;
    int k = idx / N;
    int n = idx - k * N;                       // consecutive threads -> coalesced read
    Wt[(size_t)n * K + k] = f2bf(W[(size_t)k * N + n]);
}

// ---------------------------------------------------------------------------
// LayerNorm over 1024 cols, one row per block, bf16 output.
// ---------------------------------------------------------------------------
__global__ __launch_bounds__(256) void layernorm_bf16_kernel(
    const float* __restrict__ X, const float* __restrict__ gamma,
    const float* __restrict__ beta, unsigned short* __restrict__ Y) {
    const int row = blockIdx.x;
    const float* xr = X + (size_t)row * 1024;
    float v[4], s = 0.f, ss = 0.f;
#pragma unroll
    for (int i = 0; i < 4; ++i) {
        v[i] = xr[threadIdx.x + i * 256];
        s += v[i];
        ss += v[i] * v[i];
    }
#pragma unroll
    for (int m = 1; m < 32; m <<= 1) {
        s  += __shfl_xor(s,  m, 32);
        ss += __shfl_xor(ss, m, 32);
    }
    __shared__ float red[16];
    const int wv = threadIdx.x >> 5, ln = threadIdx.x & 31;
    if (ln == 0) { red[wv] = s; red[8 + wv] = ss; }
    __syncthreads();
    s = 0.f; ss = 0.f;
#pragma unroll
    for (int w = 0; w < 8; ++w) { s += red[w]; ss += red[8 + w]; }
    const float mu   = s * (1.f / 1024.f);
    const float var  = fmaxf(ss * (1.f / 1024.f) - mu * mu, 0.f);
    const float rstd = rsqrtf(var + 1e-5f);
    unsigned short* yr = Y + (size_t)row * 1024;
#pragma unroll
    for (int i = 0; i < 4; ++i) {
        int c = threadIdx.x + i * 256;
        yr[c] = f2bf((v[i] - mu) * rstd * gamma[c] + beta[c]);
    }
}

// ---------------------------------------------------------------------------
// bf16 WMMA GEMM: C[M,N] = A[M,K] * Bt[N,K]^T
// 256 threads (8 waves, 4x2); block tile 128x128, K-step 32, LDS double buffer.
// Each wave: 32(M) x 64(N) -> 2x4 = 8 accumulators, 8 wmma per K-step.
// ---------------------------------------------------------------------------
__global__ __launch_bounds__(256) void gemm_bf16_wmma_kernel(
    const unsigned short* __restrict__ A,   // [M,K] bf16
    const unsigned short* __restrict__ Bt,  // [N,K] bf16 (pre-transposed)
    unsigned short* __restrict__ Cbf,       // [M,N] bf16 (or null)
    float* __restrict__ Cf,                 // [M,N] f32  (or null)
    const float* __restrict__ bias,         // [N] (or null)
    int M, int K, int N) {
    __shared__ __align__(32) unsigned short Asm[2][128 * 32];
    __shared__ __align__(32) unsigned short Bsm[2][128 * 32];

    const int lane = threadIdx.x & 31;
    const int wave = threadIdx.x >> 5;      // 0..7
    const int half = lane >> 4;             // 0/1
    const int l16  = lane & 15;
    const int wm   = wave >> 1;             // 0..3 (M strips of 32)
    const int wn   = wave & 1;              // 0..1 (N strips of 64)
    const int m0   = blockIdx.y * 128;
    const int n0   = blockIdx.x * 128;

    // cooperative fill mapping: thread -> (row 0..127, 16-elem segment 0/16)
    const int fr = threadIdx.x >> 1;
    const int fs = (threadIdx.x & 1) * 16;
    const unsigned short* Ag = A  + (size_t)(m0 + fr) * K + fs;
    const unsigned short* Bg = Bt + (size_t)(n0 + fr) * K + fs;

    v8f acc[2][4];
#pragma unroll
    for (int mt = 0; mt < 2; ++mt)
#pragma unroll
        for (int t = 0; t < 4; ++t) acc[mt][t] = (v8f)0.f;

    auto prefetch = [&](int buf, int kb) {
        copy32B_g2l(Ag + kb, &Asm[buf][fr * 32 + fs]);
        copy32B_g2l(Bg + kb, &Bsm[buf][fr * 32 + fs]);
    };

    int cur = 0;
    prefetch(0, 0);
    for (int kb = 0; kb < K; kb += 32) {
        wait_async_all();
        __syncthreads();                       // buf[cur] ready for everyone
        if (kb + 32 < K) prefetch(cur ^ 1, kb + 32);

        const unsigned short* Ab = &Asm[cur][0];
        const unsigned short* Bb = &Bsm[cur][0];
        Frag16 af[2];
#pragma unroll
        for (int mt = 0; mt < 2; ++mt) {
            const unsigned short* ar = Ab + (wm * 32 + mt * 16 + l16) * 32;
            af[mt].v = *(const v16bf*)(ar + half * 16);
        }
#pragma unroll
        for (int t = 0; t < 4; ++t) {
            Frag16 bfr;
            bfr.v = *(const v16bf*)(Bb + (wn * 64 + t * 16 + l16) * 32 + half * 16);
#pragma unroll
            for (int mt = 0; mt < 2; ++mt)
                acc[mt][t] = WMMA_BF16(af[mt].v, bfr.v, acc[mt][t]);
        }
        cur ^= 1;
        __syncthreads();                       // buf[cur_old] reads done before refill
    }

#pragma unroll
    for (int mt = 0; mt < 2; ++mt) {
#pragma unroll
        for (int t = 0; t < 4; ++t) {
            const int col = n0 + wn * 64 + t * 16 + l16;
#pragma unroll
            for (int r = 0; r < 8; ++r) {
                const int row = m0 + wm * 32 + mt * 16 + r + half * 8;
                const float vv = acc[mt][t][r];
                if (Cf) Cf[(size_t)row * N + col] = vv + (bias ? bias[col] : 0.f);
                else    Cbf[(size_t)row * N + col] = f2bf(vv);
            }
        }
    }
}

// ---------------------------------------------------------------------------
// Flash attention, one (b, h, 64-query-row tile) per block, 4 waves.
// Q pre-scaled by 1/8 (exact in bf16). K/V 32-row chunks staged in LDS
// (K via async copy, V transposed via scalar ds stores). Online softmax.
// ---------------------------------------------------------------------------
__global__ __launch_bounds__(128) void flash_attn_wmma_kernel(
    const unsigned short* __restrict__ Q,   // [B*Nq, 512] bf16
    const unsigned short* __restrict__ KV,  // [B*M, 1024] bf16 (k | v)
    unsigned short* __restrict__ O,         // [B*Nq, 512] bf16
    int Nq, int Mlen) {
    __shared__ __align__(32) unsigned short k_lds[32 * 64];      // [j][d]
    __shared__ __align__(32) unsigned short v_lds[64 * 32];      // [d][j] (transposed)
    __shared__ __align__(32) unsigned short p_lds[4 * 16 * 32];  // per-wave P slab

    const int b = blockIdx.z, h = blockIdx.y;
    const int row0 = blockIdx.x * 64;
    const int lane = threadIdx.x & 31;
    const int wave = threadIdx.x >> 5;
    const int half = lane >> 4;
    const int l16  = lane & 15;

    // ---- load Q fragments once, pre-scaled by 1/sqrt(64)=0.125 (exact) ----
    const int qrow = row0 + wave * 16 + l16;
    const unsigned short* qp = Q + ((size_t)(b * Nq + qrow)) * 512 + h * 64;
    Frag16 aq[2];
#pragma unroll
    for (int ks = 0; ks < 2; ++ks) {
#pragma unroll
        for (int i = 0; i < 8; ++i) {
            aq[ks].s[i]     = f2bf(bf2f(qp[ks * 32 + half * 16 + i])      * 0.125f);
            aq[ks].s[8 + i] = f2bf(bf2f(qp[ks * 32 + half * 16 + 8 + i]) * 0.125f);
        }
    }

    v8f acc[4];
#pragma unroll
    for (int t = 0; t < 4; ++t) acc[t] = (v8f)0.f;
    float mrun[8], lrun[8];
#pragma unroll
    for (int r = 0; r < 8; ++r) { mrun[r] = -1e30f; lrun[r] = 0.f; }

    const unsigned short* kbase = KV + (size_t)b * Mlen * 1024 + h * 64;
    const unsigned short* vbase = kbase + 512;

    const int jfill = threadIdx.x >> 2;           // 0..31: chunk row this thread fills
    const int dseg  = (threadIdx.x & 3) * 16;     // 0,16,32,48

    for (int jc = 0; jc < Mlen; jc += 32) {
        __syncthreads();                          // previous chunk fully consumed
        // cooperative fill: K via async copy, V transposed via scalar stores
        copy32B_g2l(kbase + (size_t)(jc + jfill) * 1024 + dseg,
                    k_lds + jfill * 64 + dseg);
        const unsigned short* vr = vbase + (size_t)(jc + jfill) * 1024 + dseg;
#pragma unroll
        for (int e = 0; e < 16; ++e)
            v_lds[(dseg + e) * 32 + jfill] = vr[e];
        wait_async_all();
        __syncthreads();

        // ---- S = (Q*scale) @ K^T for 32 columns: 2 tiles x 2 k-steps ----
        v8f sc[2];
        sc[0] = (v8f)0.f; sc[1] = (v8f)0.f;
#pragma unroll
        for (int ks = 0; ks < 2; ++ks) {
#pragma unroll
            for (int t = 0; t < 2; ++t) {
                Frag16 bk;
                bk.v = *(const v16bf*)(k_lds + (t * 16 + l16) * 64 + ks * 32 + half * 16);
                sc[t] = WMMA_BF16(aq[ks].v, bk.v, sc[t]);
            }
        }

        // ---- online softmax (row reductions inside 16-lane halves) ----
        unsigned short* pw = p_lds + wave * 512;
#pragma unroll
        for (int r = 0; r < 8; ++r) {
            float cm = fmaxf(sc[0][r], sc[1][r]);
            cm = fmaxf(cm, __shfl_xor(cm, 1, 32));
            cm = fmaxf(cm, __shfl_xor(cm, 2, 32));
            cm = fmaxf(cm, __shfl_xor(cm, 4, 32));
            cm = fmaxf(cm, __shfl_xor(cm, 8, 32));
            const float mn   = fmaxf(mrun[r], cm);
            const float corr = __expf(mrun[r] - mn);
            mrun[r] = mn;
            const float e0 = __expf(sc[0][r] - mn);
            const float e1 = __expf(sc[1][r] - mn);
            float rsum = e0 + e1;
            rsum += __shfl_xor(rsum, 1, 32);
            rsum += __shfl_xor(rsum, 2, 32);
            rsum += __shfl_xor(rsum, 4, 32);
            rsum += __shfl_xor(rsum, 8, 32);
            lrun[r] = lrun[r] * corr + rsum;
            acc[0][r] *= corr; acc[1][r] *= corr;
            acc[2][r] *= corr; acc[3][r] *= corr;
            const int prow = r + half * 8;        // C-layout row -> P row
            pw[prow * 32 + l16]      = f2bf(e0);
            pw[prow * 32 + 16 + l16] = f2bf(e1);
        }

        // ---- re-stripe P (C-layout -> A-layout) via wave-private LDS ----
        Frag16 ap;
        const unsigned short* pr = pw + l16 * 32;
        ap.v = *(const v16bf*)(pr + half * 16);

        // ---- O += P @ V (V transposed in LDS -> contiguous B frags) ----
#pragma unroll
        for (int t = 0; t < 4; ++t) {
            Frag16 bv;
            bv.v = *(const v16bf*)(v_lds + (t * 16 + l16) * 32 + half * 16);
            acc[t] = WMMA_BF16(ap.v, bv.v, acc[t]);
        }
    }

    // ---- normalize + store O (bf16) ----
    unsigned short* ob = O + ((size_t)(b * Nq + row0 + wave * 16)) * 512 + h * 64;
#pragma unroll
    for (int t = 0; t < 4; ++t) {
#pragma unroll
        for (int r = 0; r < 8; ++r) {
            const int rr = r + half * 8;
            ob[(size_t)rr * 512 + t * 16 + l16] = f2bf(acc[t][r] / lrun[r]);
        }
    }
}

// ---------------------------------------------------------------------------
// Launch
// ---------------------------------------------------------------------------
extern "C" void kernel_launch(void* const* d_in, const int* in_sizes, int n_in,
                              void* d_out, int out_size, void* d_ws, size_t ws_size,
                              hipStream_t stream) {
    (void)in_sizes; (void)n_in; (void)out_size; (void)ws_size;
    const float* x    = (const float*)d_in[0];
    const float* ctx  = (const float*)d_in[1];
    const float* g1   = (const float*)d_in[2];
    const float* b1   = (const float*)d_in[3];
    const float* g2   = (const float*)d_in[4];
    const float* b2   = (const float*)d_in[5];
    const float* Wq   = (const float*)d_in[6];
    const float* Wkv  = (const float*)d_in[7];
    const float* Wout = (const float*)d_in[8];
    const float* bout = (const float*)d_in[9];
    float* out = (float*)d_out;

    constexpr int B = 4, N = 2048, M = 2048, DIM = 1024, INNER = 512;
    constexpr int ROWS = B * N;   // 8192

    char* ws = (char*)d_ws;
    size_t off = 0;
    unsigned short* wqt   = (unsigned short*)(ws + off); off += (size_t)INNER * DIM * 2;
    unsigned short* wkvt  = (unsigned short*)(ws + off); off += (size_t)(2 * INNER) * DIM * 2;
    unsigned short* woutt = (unsigned short*)(ws + off); off += (size_t)DIM * INNER * 2;
    unsigned short* xn    = (unsigned short*)(ws + off); off += (size_t)ROWS * DIM * 2;
    unsigned short* cn    = (unsigned short*)(ws + off); off += (size_t)ROWS * DIM * 2;
    unsigned short* qb    = (unsigned short*)(ws + off); off += (size_t)ROWS * INNER * 2;
    unsigned short* kvb   = (unsigned short*)(ws + off); off += (size_t)ROWS * 2 * INNER * 2;
    unsigned short* ob    = (unsigned short*)(ws + off); off += (size_t)ROWS * INNER * 2;

    // 1) weights -> bf16, transposed
    cvt_transpose_kernel<<<(DIM * INNER + 255) / 256, 256, 0, stream>>>(Wq, wqt, DIM, INNER);
    cvt_transpose_kernel<<<(DIM * 2 * INNER + 255) / 256, 256, 0, stream>>>(Wkv, wkvt, DIM, 2 * INNER);
    cvt_transpose_kernel<<<(INNER * DIM + 255) / 256, 256, 0, stream>>>(Wout, woutt, INNER, DIM);

    // 2) layernorms -> bf16
    layernorm_bf16_kernel<<<ROWS, 256, 0, stream>>>(x, g1, b1, xn);
    layernorm_bf16_kernel<<<ROWS, 256, 0, stream>>>(ctx, g2, b2, cn);

    // 3) Q = xn @ Wq ; KV = cn @ Wkv   (128x128 tiles)
    gemm_bf16_wmma_kernel<<<dim3(INNER / 128, ROWS / 128), 256, 0, stream>>>(
        xn, wqt, qb, nullptr, nullptr, ROWS, DIM, INNER);
    gemm_bf16_wmma_kernel<<<dim3(2 * INNER / 128, ROWS / 128), 256, 0, stream>>>(
        cn, wkvt, kvb, nullptr, nullptr, ROWS, DIM, 2 * INNER);

    // 4) attention
    flash_attn_wmma_kernel<<<dim3(N / 64, 8, B), 128, 0, stream>>>(qb, kvb, ob, N, M);

    // 5) out = O @ Wout + bout  (f32)
    gemm_bf16_wmma_kernel<<<dim3(DIM / 128, ROWS / 128), 256, 0, stream>>>(
        ob, woutt, nullptr, out, bout, ROWS, INNER, DIM);
}
